// Model_65695819759990
// MI455X (gfx1250) — matrix operand order repaired and checked by reading, not compiled, not run on hardware
//
#include <hip/hip_runtime.h>
#include <hip/hip_bf16.h>
#include <cstdint>

// ---------------------------------------------------------------------------
// MI455X (gfx1250) implementation.
// Convs (657 GFLOP) -> implicit GEMM on v_wmma_f32_16x16x32_bf16 (wave32),
// A-tiles staged into LDS by the Tensor Data Mover (tensor_load_to_lds).
// Jacobi solve -> LDS-resident (256KB of the 320KB WGP LDS), 1 WG per (eq,b).
// ---------------------------------------------------------------------------

typedef __bf16 bf16_t;
typedef __attribute__((ext_vector_type(16))) __bf16 v16bf;
typedef __attribute__((ext_vector_type(8)))  __bf16 v8bf;
typedef __attribute__((ext_vector_type(8)))  float  v8f;
typedef __attribute__((ext_vector_type(4)))  unsigned int v4u;
typedef __attribute__((ext_vector_type(8)))  int v8i_t;
typedef __attribute__((ext_vector_type(4)))  int v4i_t;

static constexpr int    kBS    = 128;
static constexpr int    kG     = 32 * 32 * 32;         // 32768 cells per image
static constexpr size_t kPlane = (size_t)kBS * kG;     // 4,194,304 elements

#if __has_builtin(__builtin_amdgcn_tensor_load_to_lds)
#define USE_TDM 1
#else
#define USE_TDM 0
#endif

// ---------------------------------------------------------------------------
// TDM: 1-D contiguous row load (nelem bf16) global -> LDS. D# per ISA ch.8:
// group0 = {count=1 | flags=0, lds_addr, global_addr[31:0],
//           global_addr[56:32] | type=2<<30}
// group1 = {data_size=2B, tensor_dim0 (bits 63:48 / 79:64), tensor_dim1=1,
//           tile_dim0=nelem, tile_dim1/2=0 (unused), strides=0}
// This toolchain (clang-23 / therock-10.0) uses the 6-arg builtin form:
// (g0, g1, g2, g3, g4, cpol) — trailing groups zero for a 1-D/2-D tile.
// ---------------------------------------------------------------------------
__device__ __forceinline__ void tdm_load_row(const void* gsrc, unsigned lds_off,
                                             int nelem) {
#if USE_TDM
  unsigned long long ga = (unsigned long long)(uintptr_t)gsrc;
  v4u g0 = { 1u,                                        // count=1, user D#
             lds_off,                                   // lds_addr (bytes)
             (unsigned)(ga & 0xffffffffu),              // global_addr[31:0]
             ((unsigned)((ga >> 32) & 0x01ffffffu)) | (2u << 30) };  // type=2
  v8i_t g1 = { (int)(1u << 16),                          // data_size: 1 -> 2 bytes
               (int)(((unsigned)nelem & 0xffffu) << 16), // tensor_dim0[15:0]
               (int)((((unsigned)nelem >> 16) & 0xffffu) | (1u << 16)), // dim0 hi | tensor_dim1=1
               (int)(((unsigned)nelem & 0xffffu) << 16), // tensor_dim1 hi=0 | tile_dim0
               0, 0, 0, 0 };                             // tile_dim1/2=0, strides=0
  v4i_t z4 = { 0, 0, 0, 0 };
  v8i_t z8 = { 0, 0, 0, 0, 0, 0, 0, 0 };
  __builtin_amdgcn_tensor_load_to_lds(g0, g1, z4, z4, z8, 0);
#else
  (void)gsrc; (void)lds_off; (void)nelem;
#endif
}

__device__ __forceinline__ void wait_tensor0() {
#if USE_TDM && __has_builtin(__builtin_amdgcn_s_wait_tensorcnt)
  __builtin_amdgcn_s_wait_tensorcnt(0);
#endif
}

// ---------------------------------------------------------------------------
// NCHW fp32 -> NHWC bf16 (C = 32)
// ---------------------------------------------------------------------------
__global__ void nchw_to_nhwc(const float* __restrict__ in, bf16_t* __restrict__ out) {
  int idx = blockIdx.x * 256 + threadIdx.x;            // over 4,194,304
  int b = idx >> 15;
  int c = (idx >> 10) & 31;
  int h = (idx >> 5) & 31;
  int w = idx & 31;
  out[(size_t)(((b * 32 + h) * 32 + w) * 32) + c] = (bf16_t)in[idx];
}

// ---------------------------------------------------------------------------
// Weight repack: fp32 [O][I][3][3] -> bf16 [tap=ky*3+kx][O][I]
// (B-fragment for v_wmma: lane reads 16 contiguous K values at row N)
// ---------------------------------------------------------------------------
__global__ void pack_weights(const float* __restrict__ w, bf16_t* __restrict__ wt,
                             int O, int I) {
  int idx = blockIdx.x * 256 + threadIdx.x;
  int total = O * I * 9;
  if (idx >= total) return;
  int o = idx / (I * 9);
  int r = idx - o * I * 9;
  int i = r / 9;
  int tap = r - i * 9;                                  // w flat = (o*I+i)*9 + tap
  wt[((size_t)tap * O + o) * I + i] = (bf16_t)w[idx];
}

// ---------------------------------------------------------------------------
// 3x3 SAME conv as implicit GEMM over WMMA bf16.
//   in : [128,32,32,CIN]  NHWC bf16
//   wt : [9][COUT][CIN]   bf16
//   mode: 0 = relu, 1 = linear (fp32 NCHW out), 2 = +residual then relu
// Block = 256 thr (8 waves). Grid = (16 row-pairs, 128 batch).
// Wave (mi,ni): output row y0+mi, out-ch [ni*32,ni*32+32); 4 accums/wave so
// each B fragment feeds two WMMAs (halves weight traffic from L2).
// ---------------------------------------------------------------------------
template <int CIN, int COUT>
__global__ __launch_bounds__(256) void conv3x3_wmma(
    const bf16_t* __restrict__ in, const bf16_t* __restrict__ wt,
    const float* __restrict__ bias, const bf16_t* __restrict__ res,
    bf16_t* __restrict__ out_bf, float* __restrict__ out_f32, int mode) {
  __shared__ bf16_t smem[4 * 34 * CIN];     // rows y0-1..y0+2, cols -1..32
  const int b = blockIdx.y;
  const int y0 = blockIdx.x * 2;            // first output row of the pair
  const int tid = threadIdx.x;
  constexpr int VPC = CIN / 8;

#if USE_TDM
  {
    // Zero halo columns (cc 0,33) and out-of-range rows with plain ds stores.
    v8bf zv = {};
    for (int t = tid; t < 4 * 34; t += 256) {
      int rr = t / 34, cc = t - rr * 34;
      int h = y0 + rr - 1;
      if (cc == 0 || cc == 33 || h < 0 || h > 31) {
#pragma unroll
        for (int vv = 0; vv < VPC; ++vv)
          *(v8bf*)(smem + (size_t)(rr * 34 + cc) * CIN + vv * 8) = zv;
      }
    }
    // Interior 32-col spans are contiguous in NHWC: one 1-D TDM per valid row.
    if (tid < 32) {
#pragma unroll
      for (int rr = 0; rr < 4; ++rr) {
        int h = y0 + rr - 1;
        if (h >= 0 && h < 32)
          tdm_load_row(in + ((size_t)((b * 32 + h) * 32)) * CIN,
                       (unsigned)(uintptr_t)(smem + (size_t)(rr * 34 + 1) * CIN),
                       32 * CIN);
      }
    }
    wait_tensor0();
  }
#else
  for (int t = tid; t < 136 * VPC; t += 256) {
    int chunk = t / VPC, vv = t - chunk * VPC;
    int rr = chunk / 34, cc = chunk - rr * 34;
    int h = y0 + rr - 1, wc = cc - 1;
    v8bf val = {};
    if (h >= 0 && h < 32 && wc >= 0 && wc < 32)
      val = *(const v8bf*)(in + ((size_t)((b * 32 + h) * 32 + wc)) * CIN + vv * 8);
    *(v8bf*)(smem + (size_t)(rr * 34 + cc) * CIN + vv * 8) = val;
  }
#endif
  __syncthreads();

  const int lane = tid & 31, wave = tid >> 5;
  const int mi = wave & 1;                  // output row within pair
  const int ni = wave >> 1;
  const int nb0 = ni * 32, nb1 = ni * 32 + 16;
  if (nb0 >= COUT) return;                  // idle waves (COUT==32); no barriers below
  const int lq = lane & 15;
  const int lh = lane >> 4;

  v8f acc[2][2] = {};                       // [msub][nfrag]
  for (int tap = 0; tap < 9; ++tap) {
    const int dy = tap / 3, dx = tap - dy * 3;
    const bf16_t* wtap = wt + (size_t)tap * COUT * CIN;
    const bf16_t* brow0 = wtap + (size_t)(nb0 + lq) * CIN + lh * 16;
    const bf16_t* brow1 = wtap + (size_t)(nb1 + lq) * CIN + lh * 16;
    // pixel p = mi*32 + msub*16 + lq -> LDS row mi+dy, col (p&31)+dx
    const bf16_t* arow0 = smem + (size_t)((mi + dy) * 34 + lq + dx) * CIN;
    const bf16_t* arow1 = smem + (size_t)((mi + dy) * 34 + 16 + lq + dx) * CIN;
#pragma unroll
    for (int cs = 0; cs < CIN; cs += 32) {
      // B fragment: lanes 0-15 K=0..15, lanes 16-31 K=16..31, N=lane&15
      v16bf b0 = *(const v16bf*)(brow0 + cs);
      v16bf b1 = *(const v16bf*)(brow1 + cs);
      // A fragment (matches B): k = lh*16 + e
      v8bf a0lo = *(const v8bf*)(arow0 + cs + lh * 16);
      v8bf a0hi = *(const v8bf*)(arow0 + cs + lh * 16 + 8);
      v16bf a0 = __builtin_shufflevector(a0lo, a0hi, 0, 1, 2, 3, 4, 5, 6, 7,
                                         8, 9, 10, 11, 12, 13, 14, 15);
      acc[0][0] = __builtin_amdgcn_wmma_f32_16x16x32_bf16(false, a0, false, b0,
                                                          (short)0, acc[0][0], false, false);
      acc[0][1] = __builtin_amdgcn_wmma_f32_16x16x32_bf16(false, a0, false, b1,
                                                          (short)0, acc[0][1], false, false);
      v8bf a1lo = *(const v8bf*)(arow1 + cs + lh * 16);
      v8bf a1hi = *(const v8bf*)(arow1 + cs + lh * 16 + 8);
      v16bf a1 = __builtin_shufflevector(a1lo, a1hi, 0, 1, 2, 3, 4, 5, 6, 7,
                                         8, 9, 10, 11, 12, 13, 14, 15);
      acc[1][0] = __builtin_amdgcn_wmma_f32_16x16x32_bf16(false, a1, false, b0,
                                                          (short)0, acc[1][0], false, false);
      acc[1][1] = __builtin_amdgcn_wmma_f32_16x16x32_bf16(false, a1, false, b1,
                                                          (short)0, acc[1][1], false, false);
    }
  }

  // Epilogue: C/D layout -> n = lane&15, m = rr + 8*lh
  const int n0 = nb0 + lq;
  const int n1 = nb1 + lq;
  const float bi0 = bias[n0];
  const float bi1 = bias[n1];
  const int y = y0 + mi;
#pragma unroll
  for (int msub = 0; msub < 2; ++msub) {
#pragma unroll
    for (int rr = 0; rr < 8; ++rr) {
      const int x = msub * 16 + rr + 8 * lh;
      const size_t pix = (size_t)((b * 32 + y) * 32 + x);
      float v0 = acc[msub][0][rr] + bi0;
      float v1 = acc[msub][1][rr] + bi1;
      if (mode == 2) {
        v0 += (float)res[pix * COUT + n0];
        v1 += (float)res[pix * COUT + n1];
      }
      if (mode != 1) {
        v0 = fmaxf(v0, 0.f);
        v1 = fmaxf(v1, 0.f);
      }
      if (out_bf) {
        out_bf[pix * COUT + n0] = (bf16_t)v0;
        out_bf[pix * COUT + n1] = (bf16_t)v1;
      }
      if (out_f32) {
        out_f32[(size_t)((b * COUT + n0) * 32 + y) * 32 + x] = v0;
        out_f32[(size_t)((b * COUT + n1) * 32 + y) * 32 + x] = v1;
      }
    }
  }
}

// ---------------------------------------------------------------------------
// MLPs (512->1024->1024->3/2) + derived PDE scalars. Single workgroup.
// scal: [0]=1/ht [1]=1/hx^2 [2]=1/hy^2 [3]=c0u [4]=cdu [5]=du
//       [6]=c0v [7]=cdv [8]=dv [9]=pu2
// ---------------------------------------------------------------------------
__global__ __launch_bounds__(1024) void mlp_scalars(
    const float* __restrict__ pin1, const float* __restrict__ pin2,
    const float* __restrict__ w1a, const float* __restrict__ c1a,
    const float* __restrict__ w2a, const float* __restrict__ c2a,
    const float* __restrict__ w3a, const float* __restrict__ c3a,
    const float* __restrict__ w1b, const float* __restrict__ c1b,
    const float* __restrict__ w2b, const float* __restrict__ c2b,
    const float* __restrict__ w3b, const float* __restrict__ c3b,
    const float* __restrict__ steps0, const float* __restrict__ steps1,
    float* __restrict__ pu_out, float* __restrict__ pv_out,
    float* __restrict__ scal) {
  __shared__ float xs[512];
  __shared__ float h1[1024];
  __shared__ float h2[1024];
  __shared__ float pres[8];
  const int tid = threadIdx.x;
  for (int net = 0; net < 2; ++net) {
    const float* pin = net ? pin2 : pin1;
    const float* w1 = net ? w1b : w1a; const float* b1 = net ? c1b : c1a;
    const float* w2 = net ? w2b : w2a; const float* b2 = net ? c2b : c2a;
    const float* w3 = net ? w3b : w3a; const float* b3 = net ? c3b : c3a;
    const int nout = net ? 2 : 3;
    if (tid < 512) xs[tid] = pin[tid];
    __syncthreads();
    float s = b1[tid];
    for (int i = 0; i < 512; ++i) s = fmaf(xs[i], w1[i * 1024 + tid], s);
    h1[tid] = fmaxf(s, 0.f);
    __syncthreads();
    s = b2[tid];
    for (int i = 0; i < 1024; ++i) s = fmaf(h1[i], w2[i * 1024 + tid], s);
    h2[tid] = fmaxf(s, 0.f);
    __syncthreads();
    if (tid < nout) {
      s = b3[tid];
      for (int i = 0; i < 1024; ++i) s = fmaf(h2[i], w3[i * nout + tid], s);
      pres[net * 4 + tid] = s;
      if (net) pv_out[tid] = s; else pu_out[tid] = s;
    }
    __syncthreads();
  }
  if (tid == 0) {
    float pu0 = pres[0], pu1 = pres[1], pu2 = pres[2];
    float pv0 = pres[4], pv1 = pres[5];
    float s0 = 1.f / (1.f + __expf(-steps0[0]));
    float ht = fminf(fmaxf(s0, 0.005f), 0.2f);
    float s1 = 1.f / (1.f + __expf(-steps1[0]));
    float hx = 2.f * fminf(fmaxf(s1, 0.005f), 0.55f);
    float hy = hx;
    float iht = 1.f / ht, ihx2 = 1.f / (hx * hx), ihy2 = 1.f / (hy * hy);
    float c0u = -1.f - pu0, cdu = pu1;
    float c0v = pv0, cdv = pv1;
    float du = c0u - iht - 2.f * cdu * ihx2 - 2.f * cdu * ihy2;
    float dv = c0v - iht - 2.f * cdv * ihx2 - 2.f * cdv * ihy2;
    du = (du >= 0.f) ? fmaxf(du, 0.1f) : fminf(du, -0.1f);
    dv = (dv >= 0.f) ? fmaxf(dv, 0.1f) : fminf(dv, -0.1f);
    scal[0] = iht; scal[1] = ihx2; scal[2] = ihy2;
    scal[3] = c0u; scal[4] = cdu; scal[5] = du;
    scal[6] = c0v; scal[7] = cdv; scal[8] = dv;
    scal[9] = pu2;
  }
}

// ---------------------------------------------------------------------------
// rhs_u = pu2 + vp*up^2 ; rhs_v = -vp*up^2
// ---------------------------------------------------------------------------
__global__ void rhs_kernel(const float* __restrict__ up, const float* __restrict__ vp,
                           const float* __restrict__ scal, float* __restrict__ rhs) {
  int i = blockIdx.x * 256 + threadIdx.x;
  float a = up[i], b = vp[i];
  float q = b * a * a;
  rhs[i] = scal[9] + q;
  rhs[kPlane + i] = -q;
}

// ---------------------------------------------------------------------------
// 15 damped-Jacobi sweeps, LDS-resident double buffer (2 x 128 KB).
// Grid = 256 (eq*128 + batch). set_iv == freeze t=0 plane + x/y perimeter,
// with boundary values taken from up/vp (since iv = get_iv(up|vp)).
// ---------------------------------------------------------------------------
__global__ __launch_bounds__(1024) void jacobi_kernel(
    const float* __restrict__ upvp,   // up then vp, [2*kPlane]
    const float* __restrict__ rhs,    // rhs_u then rhs_v
    const float* __restrict__ scal,
    float* __restrict__ out) {        // u0 then v0
  extern __shared__ float sm[];
  float* wa = sm;
  float* wb = sm + kG;
  const int bid = blockIdx.x;
  const int eq = bid >> 7, b = bid & 127;
  const float* bvals = upvp + (size_t)eq * kPlane + (size_t)b * kG;
  const float* r = rhs + (size_t)eq * kPlane + (size_t)b * kG;
  const float iht = scal[0], ihx2 = scal[1], ihy2 = scal[2];
  const float c0 = eq ? scal[6] : scal[3];
  const float cd = eq ? scal[7] : scal[4];
  const float wOd = 0.7f / (eq ? scal[8] : scal[5]);
  const int tid = threadIdx.x;

  for (int k = tid; k < kG; k += 1024) {
    int t = k >> 10, x = (k >> 5) & 31, y = k & 31;
    bool bnd = (t == 0) | (x == 0) | (x == 31) | (y == 0) | (y == 31);
    wa[k] = bnd ? bvals[k] : 0.f;
  }
  __syncthreads();

  float* cur = wa;
  float* nxt = wb;
  for (int it = 0; it < 15; ++it) {
    for (int k = tid; k < kG; k += 1024) {
      int t = k >> 10, x = (k >> 5) & 31, y = k & 31;
      float wc = cur[k];
      bool bnd = (t == 0) | (x == 0) | (x == 31) | (y == 0) | (y == 31);
      if (bnd) { nxt[k] = wc; continue; }
      float ut = (t < 31) ? (cur[k + 1024] - wc) * iht : 0.f;   // fwd-diff, pad 0
      float uxx = (cur[k + 32] - 2.f * wc + cur[k - 32]) * ihx2;
      float uyy = (cur[k + 1] - 2.f * wc + cur[k - 1]) * ihy2;
      float L = c0 * wc + ut + cd * uxx + cd * uyy;             // c1 == 1
      nxt[k] = wc + wOd * (r[k] - L);
    }
    __syncthreads();
    float* tmp = cur; cur = nxt; nxt = tmp;
  }
  float* o = out + (size_t)eq * kPlane + (size_t)b * kG;
  for (int k = tid; k < kG; k += 1024) o[k] = cur[k];
}

// ---------------------------------------------------------------------------
// d_in layout (setup_inputs order, dicts flattened in insertion order):
//  0:u 1:v | 2..21 rnet1 (w_in,b_in, 4x[w1,b1,w2,b2], w_out,b_out)
//  22..41 rnet2 | 42:param_in 43:param_in2
//  44..49 pnet1 (w1,b1,w2,b2,w3,b3) | 50..55 pnet2 | 56:steps0 57:steps1
// d_out: u0 | v0 | up | vp | pu(3) | pv(2)   (float32)
// ---------------------------------------------------------------------------
extern "C" void kernel_launch(void* const* d_in, const int* in_sizes, int n_in,
                              void* d_out, int out_size, void* d_ws, size_t ws_size,
                              hipStream_t stream) {
  (void)in_sizes; (void)n_in; (void)out_size; (void)ws_size;

  const float* u = (const float*)d_in[0];
  const float* v = (const float*)d_in[1];

  // Workspace carve-up (all 16B-aligned): ~109.4 MB total
  char* ws = (char*)d_ws;
  bf16_t* act_in = (bf16_t*)(ws);                                   //  8,388,608 B
  bf16_t* bufA   = (bf16_t*)(ws + 8388608);                         // 33,554,432 B
  bf16_t* bufB   = (bf16_t*)(ws + 8388608 + 33554432);              // 33,554,432 B
  bf16_t* wpack  = (bf16_t*)(ws + 8388608 + 2ull * 33554432);       //    294,912 B
  float*  rhs    = (float*)(ws + 8388608 + 2ull * 33554432 + 294912);
  float*  scal   = (float*)(ws + 8388608 + 2ull * 33554432 + 294912 + 33554432);

  float* out    = (float*)d_out;
  float* up_out = out + 2 * kPlane;
  float* vp_out = out + 3 * kPlane;
  float* pu_out = out + 4 * kPlane;
  float* pv_out = pu_out + 3;

  // 1) MLPs + PDE scalars
  mlp_scalars<<<1, 1024, 0, stream>>>(
      (const float*)d_in[42], (const float*)d_in[43],
      (const float*)d_in[44], (const float*)d_in[45], (const float*)d_in[46],
      (const float*)d_in[47], (const float*)d_in[48], (const float*)d_in[49],
      (const float*)d_in[50], (const float*)d_in[51], (const float*)d_in[52],
      (const float*)d_in[53], (const float*)d_in[54], (const float*)d_in[55],
      (const float*)d_in[56], (const float*)d_in[57],
      pu_out, pv_out, scal);

  const dim3 cgrid(16, 128), cblk(256);

  // 2) ResNets (sequential; buffers reused)
  for (int net = 0; net < 2; ++net) {
    const int base = 2 + net * 20;
    nchw_to_nhwc<<<16384, 256, 0, stream>>>(net ? v : u, act_in);

    pack_weights<<<(36864 + 255) / 256, 256, 0, stream>>>(
        (const float*)d_in[base + 0], wpack, 128, 32);
    conv3x3_wmma<32, 128><<<cgrid, cblk, 0, stream>>>(
        act_in, wpack, (const float*)d_in[base + 1], nullptr, bufA, nullptr, 0);

    for (int blk = 0; blk < 4; ++blk) {
      const int wb = base + 2 + blk * 4;
      pack_weights<<<(147456 + 255) / 256, 256, 0, stream>>>(
          (const float*)d_in[wb + 0], wpack, 128, 128);
      conv3x3_wmma<128, 128><<<cgrid, cblk, 0, stream>>>(
          bufA, wpack, (const float*)d_in[wb + 1], nullptr, bufB, nullptr, 0);
      pack_weights<<<(147456 + 255) / 256, 256, 0, stream>>>(
          (const float*)d_in[wb + 2], wpack, 128, 128);
      conv3x3_wmma<128, 128><<<cgrid, cblk, 0, stream>>>(
          bufB, wpack, (const float*)d_in[wb + 3], bufA, bufA, nullptr, 2);
    }

    pack_weights<<<(36864 + 255) / 256, 256, 0, stream>>>(
        (const float*)d_in[base + 18], wpack, 32, 128);
    conv3x3_wmma<128, 32><<<cgrid, cblk, 0, stream>>>(
        bufA, wpack, (const float*)d_in[base + 19], nullptr, nullptr,
        net ? vp_out : up_out, 1);
  }

  // 3) RHS then LDS-resident Jacobi (writes u0|v0 into d_out)
  rhs_kernel<<<16384, 256, 0, stream>>>(up_out, vp_out, scal, rhs);

  (void)hipFuncSetAttribute(reinterpret_cast<const void*>(jacobi_kernel),
                            hipFuncAttributeMaxDynamicSharedMemorySize, 262144);
  jacobi_kernel<<<256, 1024, 262144, stream>>>(up_out, rhs, scal, out);
}
